// Classifier_38070590112093
// MI455X (gfx1250) — matrix operand order repaired and checked by reading, not compiled, not run on hardware
//
#include <hip/hip_runtime.h>

#define TDIM   64
#define HEADS  4
#define HID    32

typedef __attribute__((ext_vector_type(16))) _Float16 v16h;
typedef __attribute__((ext_vector_type(8)))  float    v8f;

// ---------- ordered-float <-> uint encoding for atomicMax ----------
__device__ __forceinline__ unsigned fenc(float x) {
  unsigned u = __float_as_uint(x);
  return (u & 0x80000000u) ? ~u : (u | 0x80000000u);
}
__device__ __forceinline__ float fdec(unsigned u) {
  unsigned b = (u & 0x80000000u) ? (u & 0x7FFFFFFFu) : ~u;
  return __uint_as_float(b);
}

__device__ __forceinline__ void edge_nodes(const int* __restrict__ ei, int NE,
                                           int e, int& s, int& d) {
  if (e < NE) { s = ei[e]; d = ei[NE + e]; }
  else        { s = e - NE; d = e - NE; }      // self loop
}

// ---------- utility: zero a float/uint region ----------
__global__ void k_zero(float* __restrict__ p, long n) {
  long i = (long)blockIdx.x * blockDim.x + threadIdx.x;
  if (i < n) p[i] = 0.0f;
}

// ---------- weight convert + transpose to f16 ----------
// Wt1: [128 cols][64 k]  from W1 [64][128]
// Wt2: [32 cols][128 k]  from W2 [128][32]
__global__ void k_convw(const float* __restrict__ W1, const float* __restrict__ W2,
                        _Float16* __restrict__ Wt1, _Float16* __restrict__ Wt2) {
  int t = blockIdx.x * blockDim.x + threadIdx.x;
  if (t < 128 * 64) {
    int c = t / 64, k = t - c * 64;
    Wt1[t] = (_Float16)W1[k * 128 + c];
  } else if (t < 128 * 64 + 32 * 128) {
    int u = t - 128 * 64;
    int c = u / 128, k = u - c * 128;
    Wt2[u] = (_Float16)W2[k * 32 + c];
  }
}

// ---------- GEMM1: H1[N,128] = x[N,64] @ W1 (via WMMA f16) ----------
__global__ void k_gemm1(const float* __restrict__ x, const _Float16* __restrict__ Wt,
                        float* __restrict__ H, int N) {
  __shared__ __align__(16) _Float16 sA[16 * 64];
  int r0 = blockIdx.x * 16;
  if (r0 >= N) return;
  int t = threadIdx.x;                       // 256 threads = 8 waves
  const float4* xg = reinterpret_cast<const float4*>(x + (size_t)r0 * 64);
  float4 v = xg[t];                          // 256 float4 = 16x64 tile
  _Float16* sp = &sA[t * 4];
  sp[0] = (_Float16)v.x; sp[1] = (_Float16)v.y;
  sp[2] = (_Float16)v.z; sp[3] = (_Float16)v.w;
  __syncthreads();

  int lane = t & 31, wave = t >> 5;
  int row  = lane & 15;
  int kb   = (lane < 16) ? 0 : 8;            // unused: A loads use koff as well
  int koff = (lane < 16) ? 0 : 16;           // B: K offset per lane group
  v16h a0, a1, b0, b1;
  uint4* a0p = reinterpret_cast<uint4*>(&a0);
  uint4* a1p = reinterpret_cast<uint4*>(&a1);
  a0p[0] = *reinterpret_cast<const uint4*>(&sA[row * 64 + koff]);
  a0p[1] = *reinterpret_cast<const uint4*>(&sA[row * 64 + koff + 8]);
  a1p[0] = *reinterpret_cast<const uint4*>(&sA[row * 64 + 32 + koff]);
  a1p[1] = *reinterpret_cast<const uint4*>(&sA[row * 64 + 32 + koff + 8]);

  int col = wave * 16 + (lane & 15);
  const _Float16* wp = Wt + col * 64;
  uint4* b0p = reinterpret_cast<uint4*>(&b0);
  uint4* b1p = reinterpret_cast<uint4*>(&b1);
  b0p[0] = *reinterpret_cast<const uint4*>(wp + koff);
  b0p[1] = *reinterpret_cast<const uint4*>(wp + koff + 8);
  b1p[0] = *reinterpret_cast<const uint4*>(wp + 32 + koff);
  b1p[1] = *reinterpret_cast<const uint4*>(wp + 32 + koff + 8);

  v8f c = {};
  c = __builtin_amdgcn_wmma_f32_16x16x32_f16(false, a0, false, b0, (short)0, c, false, false);
  c = __builtin_amdgcn_wmma_f32_16x16x32_f16(false, a1, false, b1, (short)0, c, false, false);

  int moff = (lane < 16) ? 0 : 8;
  float* op = H + (size_t)(r0 + moff) * 128 + wave * 16 + (lane & 15);
#pragma unroll
  for (int r = 0; r < 8; ++r) op[(size_t)r * 128] = c[r];
}

// ---------- GEMM2: H2[N,32] = h1[N,128] @ W2 (via WMMA f16) ----------
__global__ void k_gemm2(const float* __restrict__ h1, const _Float16* __restrict__ Wt,
                        float* __restrict__ H2, int N) {
  __shared__ __align__(16) _Float16 sA[16 * 128];
  int r0 = blockIdx.x * 16;
  if (r0 >= N) return;
  int t = threadIdx.x;                       // 64 threads = 2 waves
  const float4* hg = reinterpret_cast<const float4*>(h1 + (size_t)r0 * 128);
#pragma unroll
  for (int i = 0; i < 8; ++i) {
    float4 v = hg[t + i * 64];
    _Float16* sp = &sA[(t + i * 64) * 4];
    sp[0] = (_Float16)v.x; sp[1] = (_Float16)v.y;
    sp[2] = (_Float16)v.z; sp[3] = (_Float16)v.w;
  }
  __syncthreads();

  int lane = t & 31, wave = t >> 5;
  int row  = lane & 15;
  int kb   = (lane < 16) ? 0 : 8;
  int koff = (lane < 16) ? 0 : 16;
  int col  = wave * 16 + (lane & 15);
  const _Float16* wp = Wt + col * 128;

  v8f c = {};
#pragma unroll
  for (int kk = 0; kk < 128; kk += 32) {
    v16h a, b;
    uint4* ap = reinterpret_cast<uint4*>(&a);
    ap[0] = *reinterpret_cast<const uint4*>(&sA[row * 128 + kk + koff]);
    ap[1] = *reinterpret_cast<const uint4*>(&sA[row * 128 + kk + koff + 8]);
    uint4* bp = reinterpret_cast<uint4*>(&b);
    bp[0] = *reinterpret_cast<const uint4*>(wp + kk + koff);
    bp[1] = *reinterpret_cast<const uint4*>(wp + kk + koff + 8);
    c = __builtin_amdgcn_wmma_f32_16x16x32_f16(false, a, false, b, (short)0, c, false, false);
  }
  int moff = (lane < 16) ? 0 : 8;
  float* op = H2 + (size_t)(r0 + moff) * 32 + wave * 16 + (lane & 15);
#pragma unroll
  for (int r = 0; r < 8; ++r) op[(size_t)r * 32] = c[r];
}

// ---------- per-node attention logits ----------
__global__ void k_alpha(const float* __restrict__ H, const float* __restrict__ avs,
                        const float* __restrict__ avd, float* __restrict__ As,
                        float* __restrict__ Ad, int N, int heads, int dim) {
  int tid = blockIdx.x * blockDim.x + threadIdx.x;
  if (tid >= N * heads) return;
  int n = tid / heads, h = tid - n * heads;
  const float* hp = H + (size_t)n * heads * dim + h * dim;
  const float* p1 = avs + h * dim;
  const float* p2 = avd + h * dim;
  float s1 = 0.f, s2 = 0.f;
  for (int j = 0; j < dim; ++j) { float v = hp[j]; s1 += v * p1[j]; s2 += v * p2[j]; }
  As[tid] = s1; Ad[tid] = s2;
}

// ---------- edge pass 1: segment max (ordered-uint atomicMax) ----------
__global__ void k_edge_max(const int* __restrict__ ei, int NE, int Etot, int heads,
                           const float* __restrict__ As, const float* __restrict__ Ad,
                           unsigned* __restrict__ Menc) {
  int t = blockIdx.x * blockDim.x + threadIdx.x;
  if (t >= Etot * heads) return;
  int e = t / heads, h = t - e * heads;
  int s, d; edge_nodes(ei, NE, e, s, d);
  float ev = As[s * heads + h] + Ad[d * heads + h];
  ev = ev > 0.f ? ev : 0.2f * ev;
  atomicMax(&Menc[d * heads + h], fenc(ev));
}

// ---------- edge pass 2: segment sum of exp ----------
__global__ void k_edge_sum(const int* __restrict__ ei, int NE, int Etot, int heads,
                           const float* __restrict__ As, const float* __restrict__ Ad,
                           const unsigned* __restrict__ Menc, float* __restrict__ S) {
  int t = blockIdx.x * blockDim.x + threadIdx.x;
  if (t >= Etot * heads) return;
  int e = t / heads, h = t - e * heads;
  int s, d; edge_nodes(ei, NE, e, s, d);
  int idx = d * heads + h;
  float ev = As[s * heads + h] + Ad[idx];
  ev = ev > 0.f ? ev : 0.2f * ev;
  atomicAdd(&S[idx], __expf(ev - fdec(Menc[idx])));
}

// ---------- edge pass 3: alpha-weighted scatter-add of messages ----------
__global__ void k_edge_agg(const int* __restrict__ ei, int NE, int Etot, int heads, int dim,
                           const float* __restrict__ As, const float* __restrict__ Ad,
                           const unsigned* __restrict__ Menc, const float* __restrict__ S,
                           const float* __restrict__ H, float* __restrict__ ACC) {
  int t = blockIdx.x * blockDim.x + threadIdx.x;
  if (t >= Etot * heads) return;
  int e = t / heads, h = t - e * heads;
  int s, d; edge_nodes(ei, NE, e, s, d);
  const float* hs = H + (size_t)s * heads * dim + h * dim;
  __builtin_prefetch(hs, 0, 0);              // global_prefetch_b8 of gathered row
  int idx = d * heads + h;
  float ev = As[s * heads + h] + Ad[idx];
  ev = ev > 0.f ? ev : 0.2f * ev;
  float alpha = __expf(ev - fdec(Menc[idx])) / (S[idx] + 1e-16f);
  float* ap = ACC + (size_t)d * heads * dim + h * dim;
#pragma unroll
  for (int j = 0; j < 32; j += 4) {
    float4 v = *reinterpret_cast<const float4*>(hs + j);
    atomicAdd(ap + j + 0, v.x * alpha);
    atomicAdd(ap + j + 1, v.y * alpha);
    atomicAdd(ap + j + 2, v.z * alpha);
    atomicAdd(ap + j + 3, v.w * alpha);
  }
}

// ---------- bias + ReLU in place ----------
__global__ void k_bias_relu(float* __restrict__ A, const float* __restrict__ b,
                            int C, long total) {
  long i = (long)blockIdx.x * blockDim.x + threadIdx.x;
  if (i >= total) return;
  int c = (int)(i % C);
  float v = A[i] + b[c];
  A[i] = v > 0.f ? v : 0.f;
}

// ---------- global mean pool (sums + counts) ----------
__global__ void k_pool(const float* __restrict__ H2, const int* __restrict__ batch,
                       float* __restrict__ GS, float* __restrict__ GC, int N) {
  int n = blockIdx.x * blockDim.x + threadIdx.x;
  if (n >= N) return;
  int g = batch[n];
  atomicAdd(&GC[g], 1.0f);
  const float* hp = H2 + (size_t)n * 32;
#pragma unroll
  for (int j = 0; j < 32; ++j) atomicAdd(&GS[g * 32 + j], hp[j]);
}

// ---------- classifier head: sigmoid(pooled @ Wc + bc) ----------
__global__ void k_final(const float* __restrict__ GS, const float* __restrict__ GC,
                        const float* __restrict__ Wc, const float* __restrict__ bc,
                        float* __restrict__ out, int G) {
  int g = blockIdx.x * blockDim.x + threadIdx.x;
  if (g >= G) return;
  float cnt = fmaxf(GC[g], 1.0f);
  float s = bc[0];
#pragma unroll
  for (int j = 0; j < 32; ++j) s += (GS[g * 32 + j] / cnt) * Wc[j];
  out[g] = 1.0f / (1.0f + __expf(-s));
}

extern "C" void kernel_launch(void* const* d_in, const int* in_sizes, int n_in,
                              void* d_out, int out_size, void* d_ws, size_t ws_size,
                              hipStream_t stream) {
  const float* x      = (const float*)d_in[0];
  const int*   ei     = (const int*)d_in[1];
  const int*   batch  = (const int*)d_in[2];
  const float* W1     = (const float*)d_in[3];
  const float* a_src1 = (const float*)d_in[4];
  const float* a_dst1 = (const float*)d_in[5];
  const float* b1     = (const float*)d_in[6];
  const float* W2     = (const float*)d_in[7];
  const float* a_src2 = (const float*)d_in[8];
  const float* a_dst2 = (const float*)d_in[9];
  const float* b2     = (const float*)d_in[10];
  const float* Wc     = (const float*)d_in[11];
  const float* bc     = (const float*)d_in[12];
  float* out = (float*)d_out;

  const int N    = in_sizes[0] / TDIM;   // 100000
  const int NE   = in_sizes[1] / 2;      // 1600000
  const int Etot = NE + N;               // with self loops
  const int G    = out_size;             // 1024

  // ---- workspace layout (256B aligned slices) ----
  char* base = (char*)d_ws;
  size_t off = 0;
  auto carve = [&](size_t bytes) -> char* {
    char* p = base + off;
    off += (bytes + 255) & ~(size_t)255;
    return p;
  };
  _Float16* Wt1 = (_Float16*)carve(128 * 64 * sizeof(_Float16));
  _Float16* Wt2 = (_Float16*)carve(32 * 128 * sizeof(_Float16));
  float*    H1  = (float*)carve((size_t)N * 128 * sizeof(float));   // reused below
  float*    ACC1= (float*)carve((size_t)N * 128 * sizeof(float));
  float*    A1s = (float*)carve((size_t)N * HEADS * sizeof(float));
  float*    A1d = (float*)carve((size_t)N * HEADS * sizeof(float));
  unsigned* M1  = (unsigned*)carve((size_t)N * HEADS * sizeof(unsigned));
  float*    S1  = (float*)carve((size_t)N * HEADS * sizeof(float));
  float*    A2s = (float*)carve((size_t)N * sizeof(float));
  float*    A2d = (float*)carve((size_t)N * sizeof(float));
  unsigned* M2  = (unsigned*)carve((size_t)N * sizeof(unsigned));
  float*    S2  = (float*)carve((size_t)N * sizeof(float));
  float*    GS  = (float*)carve((size_t)G * 32 * sizeof(float));
  float*    GC  = (float*)carve((size_t)G * sizeof(float));
  // H1 region is dead after layer-1 aggregation: alias layer-2 buffers into it.
  float*    H2   = H1;
  float*    ACC2 = H1 + (size_t)N * 32;

  const int TB = 256;
  auto blocks = [](long n, int tb) { return (unsigned)((n + tb - 1) / tb); };

  // weights -> f16 transposed
  k_convw<<<blocks(128 * 64 + 32 * 128, TB), TB, 0, stream>>>(W1, W2, Wt1, Wt2);

  // ---- layer 1 ----
  k_gemm1<<<blocks((long)N, 16), 256, 0, stream>>>(x, Wt1, H1, N);
  k_zero<<<blocks((long)N * 128, TB), TB, 0, stream>>>(ACC1, (long)N * 128);
  k_zero<<<blocks((long)N * HEADS, TB), TB, 0, stream>>>((float*)M1, (long)N * HEADS); // 0u == encoded minimum
  k_zero<<<blocks((long)N * HEADS, TB), TB, 0, stream>>>(S1, (long)N * HEADS);
  k_alpha<<<blocks((long)N * HEADS, TB), TB, 0, stream>>>(H1, a_src1, a_dst1, A1s, A1d, N, HEADS, HID);
  k_edge_max<<<blocks((long)Etot * HEADS, TB), TB, 0, stream>>>(ei, NE, Etot, HEADS, A1s, A1d, M1);
  k_edge_sum<<<blocks((long)Etot * HEADS, TB), TB, 0, stream>>>(ei, NE, Etot, HEADS, A1s, A1d, M1, S1);
  k_edge_agg<<<blocks((long)Etot * HEADS, TB), TB, 0, stream>>>(ei, NE, Etot, HEADS, HID,
                                                               A1s, A1d, M1, S1, H1, ACC1);
  k_bias_relu<<<blocks((long)N * 128, TB), TB, 0, stream>>>(ACC1, b1, 128, (long)N * 128);

  // ---- layer 2 ----
  k_gemm2<<<blocks((long)N, 16), 64, 0, stream>>>(ACC1, Wt2, H2, N);
  k_zero<<<blocks((long)N * 32, TB), TB, 0, stream>>>(ACC2, (long)N * 32);
  k_zero<<<blocks((long)N, TB), TB, 0, stream>>>((float*)M2, (long)N);
  k_zero<<<blocks((long)N, TB), TB, 0, stream>>>(S2, (long)N);
  k_alpha<<<blocks((long)N, TB), TB, 0, stream>>>(H2, a_src2, a_dst2, A2s, A2d, N, 1, HID);
  k_edge_max<<<blocks((long)Etot, TB), TB, 0, stream>>>(ei, NE, Etot, 1, A2s, A2d, M2);
  k_edge_sum<<<blocks((long)Etot, TB), TB, 0, stream>>>(ei, NE, Etot, 1, A2s, A2d, M2, S2);
  k_edge_agg<<<blocks((long)Etot, TB), TB, 0, stream>>>(ei, NE, Etot, 1, HID,
                                                        A2s, A2d, M2, S2, H2, ACC2);
  k_bias_relu<<<blocks((long)N * 32, TB), TB, 0, stream>>>(ACC2, b2, 32, (long)N * 32);

  // ---- pool + classify ----
  k_zero<<<blocks((long)G * 32, TB), TB, 0, stream>>>(GS, (long)G * 32);
  k_zero<<<blocks((long)G, TB), TB, 0, stream>>>(GC, (long)G);
  k_pool<<<blocks((long)N, TB), TB, 0, stream>>>(ACC2, batch, GS, GC, N);
  k_final<<<blocks((long)G, TB), TB, 0, stream>>>(GS, GC, Wc, bc, out, G);
}
